// HandwritingLNNAttention_74981539054060
// MI455X (gfx1250) — matrix operand order repaired and checked by reading, not compiled, not run on hardware
//
#include <hip/hip_runtime.h>
#include <hip/hip_bf16.h>

// ---------------- problem constants ----------------
#define U_    128
#define IN_   6
#define B_    1024
#define T_    96
#define M_    64
#define C_    100
#define NUNF  6
#define BB    8          // batches per workgroup in the LTC kernel
#define EPSF  1e-8f

typedef __attribute__((ext_vector_type(16))) __bf16 v16bf;
typedef __attribute__((ext_vector_type(8)))  __bf16 v8bf;
typedef __attribute__((ext_vector_type(8)))  float  v8f;
typedef __attribute__((ext_vector_type(4)))  int    i32x4;

#define SHUF16(lo, hi) __builtin_shufflevector((lo), (hi), 0,1,2,3,4,5,6,7,8,9,10,11,12,13,14,15)

// ---------------- workspace layout (bytes, 16B aligned) ----------------
#define WS_SYN      0u          // uint2[U*U]      bf16x4-packed recurrent synapses  (131072)
#define WS_SENS     131072u     // float4[6*U]     sensory synapses                  (12288)
#define WS_RSUM     143360u     // float[2*U]      hoisted 0.5*sum(we), 0.5*sum(wp)  (1024)
#define WS_SSUM     144384u     // float[2*U]      sensory hoisted sums              (1024)
#define WS_NEUR     145408u     // float[3*U]      cm_t, gleak, gleak*vleak          (1536)
#define WS_AW1T     147456u     // bf16[32*64]     aw1 transposed (NxK)              (4096)
#define WS_CW1T     151552u     // bf16[128*64]    cw1 transposed (NxK)              (16384)
#define WS_CW2T     167936u     // bf16[128*128]   cw2 transposed+zero-padded (NxK)  (32768)
#define WS_SCORES   200704u     // float[B*T]                                        (393216)
#define WS_CTX      593920u     // bf16[B*64]                                        (131072)
#define WS_H2       724992u     // bf16[B*128]                                       (262144)
#define WS_OUTS     987136u     // bf16[B*T*64]    motor outputs                     (12582912)

// ---------------- gfx1250 feature guards ----------------
#if defined(__HIP_DEVICE_COMPILE__) && __has_builtin(__builtin_amdgcn_global_load_async_to_lds_b128)
#define HAS_ASYNC_LDS 1
#else
#define HAS_ASYNC_LDS 0
#endif

__device__ __forceinline__ void copy16_to_lds(const uint4* g, uint4* l) {
#if HAS_ASYNC_LDS
  // builtin signature (from hipcc diagnostic): (i32x4 AS1*, i32x4 AS3*, imm offset, imm cpol)
  __builtin_amdgcn_global_load_async_to_lds_b128(
      (__attribute__((address_space(1))) i32x4*)g,
      (__attribute__((address_space(3))) i32x4*)l, 0, 0);
#else
  *l = *g;
#endif
}

__device__ __forceinline__ void async_lds_wait0() {
#if HAS_ASYNC_LDS && __has_builtin(__builtin_amdgcn_s_wait_asynccnt)
  __builtin_amdgcn_s_wait_asynccnt(0);
#endif
}

__device__ __forceinline__ float fast_tanh(float x) {
#if defined(__HIP_DEVICE_COMPILE__) && __has_builtin(__builtin_amdgcn_tanhf)
  return __builtin_amdgcn_tanhf(x);           // native v_tanh_f32 on gfx1250
#elif defined(__HIP_DEVICE_COMPILE__) && __has_builtin(__builtin_amdgcn_tanh_f32)
  return __builtin_amdgcn_tanh_f32(x);
#else
  float e = __expf(2.f * x);                  // inf-safe: 1 - 2/(e+1)
  return 1.f - 2.f * __builtin_amdgcn_rcpf(e + 1.f);
#endif
}

__device__ __forceinline__ float softplusf(float x) {
  return fmaxf(x, 0.f) + log1pf(__expf(-fabsf(x)));
}
__device__ __forceinline__ unsigned f2bf_bits(float f) {   // RNE f32 -> bf16 bits
  unsigned u = __float_as_uint(f);
  u += 0x7FFFu + ((u >> 16) & 1u);
  return u >> 16;
}
__device__ __forceinline__ float bf_lo(unsigned u) { return __uint_as_float(u << 16); }
__device__ __forceinline__ float bf_hi(unsigned u) { return __uint_as_float(u & 0xFFFF0000u); }

// ================= K0a: pack recurrent synapses (bf16x4 per synapse) =================
__global__ void k0_syn(const float* __restrict__ sigma, const float* __restrict__ mu,
                       const float* __restrict__ w, const float* __restrict__ erev,
                       const float* __restrict__ mask, uint2* __restrict__ syn) {
  int idx = blockIdx.x * blockDim.x + threadIdx.x;
  if (idx >= U_ * U_) return;
  float sg = sigma[idx];
  float wp = softplusf(w[idx]) * mask[idx];
  float we = wp * erev[idx];
  // sigmoid((v-mu)*sigma) = 0.5*tanh(0.5*sigma*v - 0.5*sigma*mu) + 0.5  (const part hoisted)
  unsigned sh = f2bf_bits(0.5f * sg);
  unsigned cn = f2bf_bits(-0.5f * sg * mu[idx]);
  unsigned a  = f2bf_bits(0.5f * we);
  unsigned b  = f2bf_bits(0.5f * wp);
  syn[idx] = make_uint2(sh | (cn << 16), a | (b << 16));
}

// ================= K0b: sensory synapses (f32x4) =================
__global__ void k0_sens(const float* __restrict__ ss, const float* __restrict__ smu,
                        const float* __restrict__ sw, const float* __restrict__ serev,
                        const float* __restrict__ smask, float4* __restrict__ sens) {
  int idx = blockIdx.x * blockDim.x + threadIdx.x;
  if (idx >= IN_ * U_) return;
  float sg  = ss[idx];
  float swp = softplusf(sw[idx]) * smask[idx];
  float swe = swp * serev[idx];
  sens[idx] = make_float4(0.5f * sg, -0.5f * sg * smu[idx], 0.5f * swe, 0.5f * swp);
}

// ================= K0c: per-neuron constants + hoisted half-sums =================
__global__ void k0_misc(const float* __restrict__ w, const float* __restrict__ erev,
                        const float* __restrict__ mask,
                        const float* __restrict__ sw, const float* __restrict__ serev,
                        const float* __restrict__ smask,
                        const float* __restrict__ gleak, const float* __restrict__ vleak,
                        const float* __restrict__ cm,
                        float* __restrict__ rsum, float* __restrict__ ssum,
                        float* __restrict__ neur) {
  int j = threadIdx.x;
  if (j >= U_) return;
  float rn = 0.f, rd = 0.f;
  for (int i = 0; i < U_; ++i) {
    float wp = softplusf(w[i * U_ + j]) * mask[i * U_ + j];
    rd += wp;
    rn += wp * erev[i * U_ + j];
  }
  rsum[j] = 0.5f * rn;  rsum[U_ + j] = 0.5f * rd;
  float sn = 0.f, sd = 0.f;
  for (int i = 0; i < IN_; ++i) {
    float wp = softplusf(sw[i * U_ + j]) * smask[i * U_ + j];
    sd += wp;
    sn += wp * serev[i * U_ + j];
  }
  ssum[j] = 0.5f * sn;  ssum[U_ + j] = 0.5f * sd;
  float gl = softplusf(gleak[j]);
  neur[j]          = softplusf(cm[j]) * (float)NUNF;  // cm_t
  neur[U_ + j]     = gl;
  neur[2 * U_ + j] = gl * vleak[j];
}

// ================= K0d: transpose GEMM weights to bf16 NxK =================
__global__ void k0_wt(const float* __restrict__ aw1, const float* __restrict__ cw1,
                      const float* __restrict__ cw2,
                      __bf16* __restrict__ aw1T, __bf16* __restrict__ cw1T,
                      __bf16* __restrict__ cw2T) {
  int idx = blockIdx.x * blockDim.x + threadIdx.x;
  if (idx < 32 * 64)  { int n = idx / 64, k = idx % 64;  aw1T[idx] = (__bf16)aw1[k * 32 + n]; }
  if (idx < 128 * 64) { int n = idx / 64, k = idx % 64;  cw1T[idx] = (__bf16)cw1[k * 128 + n]; }
  if (idx < 128 * 128){ int n = idx / 128, k = idx % 128;
                        cw2T[idx] = (__bf16)((n < C_) ? cw2[k * C_ + n] : 0.f); }
}

// ================= K1: LTC scan (the 9.7e9-tanh hot loop) =================
// 1024 threads = 8 batches x (4 waves of 32 lanes) ; lane owns postsynaptic neuron j.
// 148KB LDS per workgroup: synapses (bf16x4, conflict-free ds_load_b64),
// sensory table, double-buffered membrane state v.
__global__ void __launch_bounds__(1024, 1)
k1_ltc(const float* __restrict__ x,
       const float* __restrict__ ln_g, const float* __restrict__ ln_b,
       const float* __restrict__ in_w, const float* __restrict__ in_b,
       const float* __restrict__ out_w, const float* __restrict__ out_b,
       const uint2* __restrict__ syn_g, const float4* __restrict__ sens_g,
       const float* __restrict__ rsum, const float* __restrict__ ssum,
       const float* __restrict__ neur, __bf16* __restrict__ outs) {
  __shared__ uint2  s_syn[U_ * U_];      // 131072 B
  __shared__ float4 s_sens[IN_ * U_];    //  12288 B
  __shared__ float  s_v[2][BB][U_];      //   8192 B

  int tid = threadIdx.x;
  // async stage weights into LDS (b128 per lane)
  {
    const uint4* src = (const uint4*)syn_g;
    uint4* dst = (uint4*)s_syn;
    for (int k = tid; k < (U_ * U_) / 2; k += 1024) copy16_to_lds(src + k, dst + k);
    const uint4* src2 = (const uint4*)sens_g;
    uint4* dst2 = (uint4*)s_sens;
    for (int k = tid; k < IN_ * U_; k += 1024) copy16_to_lds(src2 + k, dst2 + k);
  }
  int bl = tid >> 7;          // local batch 0..7
  int j  = tid & (U_ - 1);    // postsynaptic neuron
  int b  = blockIdx.x * BB + bl;

  float rs_n = rsum[j],      rs_d = rsum[U_ + j];
  float ss_n = ssum[j],      ss_d = ssum[U_ + j];
  float cmt  = neur[j],      gl = neur[U_ + j], glv = neur[2 * U_ + j];
  float g6[IN_], b6[IN_];
#pragma unroll
  for (int i = 0; i < IN_; ++i) {         // fold LN affine + input affine
    g6[i] = ln_g[i] * in_w[i];
    b6[i] = ln_b[i] * in_w[i] + in_b[i];
  }
  float ow = 0.f, ob = 0.f;
  if (j < M_) { ow = out_w[j]; ob = out_b[j]; }

  float v = 0.f;
  s_v[0][bl][j] = 0.f;
  async_lds_wait0();
  __syncthreads();
  int cur = 0;

  const float* xb = x + (size_t)b * T_ * IN_;
  for (int t = 0; t < T_; ++t) {
    // LayerNorm over the 6 features
    float xv[IN_], mean = 0.f;
#pragma unroll
    for (int i = 0; i < IN_; ++i) { xv[i] = xb[t * IN_ + i]; mean += xv[i]; }
    mean *= (1.f / IN_);
    float var = 0.f;
#pragma unroll
    for (int i = 0; i < IN_; ++i) { float d = xv[i] - mean; var += d * d; }
    var *= (1.f / IN_);
    float inv = rsqrtf(var + 1e-5f);
    // sensory synapse currents (const parts pre-hoisted into ss_n/ss_d)
    float wn = ss_n, wd = ss_d;
#pragma unroll
    for (int i = 0; i < IN_; ++i) {
      float inp = fmaf((xv[i] - mean) * inv, g6[i], b6[i]);
      float4 sv = s_sens[i * U_ + j];
      float tt = fast_tanh(fmaf(sv.x, inp, sv.y));
      wn = fmaf(sv.z, tt, wn);
      wd = fmaf(sv.w, tt, wd);
    }
    // 6 ODE unfolds; 128-wide presynaptic reduction per lane
    for (int u = 0; u < NUNF; ++u) {
      float num = 0.f, den = 0.f;
#pragma unroll 4
      for (int i = 0; i < U_; ++i) {
        uint2 uu = s_syn[i * U_ + j];          // conflict-free ds_load_b64
        float vi = s_v[cur][bl][i];            // LDS broadcast
        float tt = fast_tanh(fmaf(bf_lo(uu.x), vi, bf_hi(uu.x)));
        num = fmaf(bf_lo(uu.y), tt, num);
        den = fmaf(bf_hi(uu.y), tt, den);
      }
      float nn = fmaf(cmt, v, glv + num + rs_n + wn);
      float dd = cmt + gl + den + rs_d + wd;
      v = nn / (dd + EPSF);
      s_v[cur ^ 1][bl][j] = v;
      __syncthreads();
      cur ^= 1;
    }
    if (j < M_) outs[((size_t)b * T_ + t) * M_ + j] = (__bf16)fmaf(v, ow, ob);
  }
}

// ================= K2: attention MLP via WMMA, fused relu+aw2 dot =================
// scores[B*T] = relu(outs @ aw1 + ab1) @ aw2 ; one wave per 16 rows, N=32 in 2 tiles.
__global__ void __launch_bounds__(256)
k2_attn(const __bf16* __restrict__ outs, const __bf16* __restrict__ aw1T,
        const float* __restrict__ ab1, const float* __restrict__ aw2,
        float* __restrict__ scores) {
  int lane = threadIdx.x & 31;
  int wid  = (blockIdx.x * blockDim.x + threadIdx.x) >> 5;
  int m0   = wid * 16;
  int half = lane >> 4, ln = lane & 15;

  const v8bf* A  = (const v8bf*)(outs + (size_t)(m0 + ln) * M_);
  const v8bf* B0 = (const v8bf*)(aw1T + (size_t)ln * 64);
  const v8bf* B1 = (const v8bf*)(aw1T + (size_t)(16 + ln) * 64);
  v8f acc0 = {}, acc1 = {};
#pragma unroll
  for (int kk = 0; kk < 2; ++kk) {
    v16bf a  = SHUF16(A[kk * 4 + half], A[kk * 4 + half + 2]);
    v16bf b0 = SHUF16(B0[kk * 4 + half], B0[kk * 4 + half + 2]);
    v16bf b1 = SHUF16(B1[kk * 4 + half], B1[kk * 4 + half + 2]);
    acc0 = __builtin_amdgcn_wmma_f32_16x16x32_bf16(false, a, false, b0, (short)0, acc0, false, false);
    acc1 = __builtin_amdgcn_wmma_f32_16x16x32_bf16(false, a, false, b1, (short)0, acc1, false, false);
  }
  float w0 = aw2[ln], w1 = aw2[16 + ln];
  float z0 = ab1[ln], z1 = ab1[16 + ln];
#pragma unroll
  for (int r = 0; r < 8; ++r) {
    float p = fmaxf(acc0[r] + z0, 0.f) * w0 + fmaxf(acc1[r] + z1, 0.f) * w1;
    p += __shfl_xor(p, 1, 16);
    p += __shfl_xor(p, 2, 16);
    p += __shfl_xor(p, 4, 16);
    p += __shfl_xor(p, 8, 16);
    if (ln == 0) scores[m0 + r + 8 * half] = p;
  }
}

// ================= K3: softmax over T + attention pooling =================
__global__ void k3_pool(const float* __restrict__ scores, const __bf16* __restrict__ outs,
                        __bf16* __restrict__ ctx) {
  __shared__ float sh[T_];
  __shared__ float s_inv;
  int b = blockIdx.x, tid = threadIdx.x;
  if (tid < T_) sh[tid] = scores[b * T_ + tid];
  __syncthreads();
  if (tid == 0) {
    float mx = sh[0];
    for (int t = 1; t < T_; ++t) mx = fmaxf(mx, sh[t]);
    float s = 0.f;
    for (int t = 0; t < T_; ++t) { float e = __expf(sh[t] - mx); sh[t] = e; s += e; }
    s_inv = 1.f / s;
  }
  __syncthreads();
  if (tid < M_) {
    float acc = 0.f;
    for (int t = 0; t < T_; ++t)
      acc += sh[t] * (float)outs[((size_t)b * T_ + t) * M_ + tid];
    ctx[b * M_ + tid] = (__bf16)(acc * s_inv);
  }
}

// ================= K4: classifier layer 1 (ctx @ cw1 + cb1, relu) via WMMA =================
__global__ void __launch_bounds__(256)
k4_fc1(const __bf16* __restrict__ ctx, const __bf16* __restrict__ cw1T,
       const float* __restrict__ cb1, __bf16* __restrict__ h2) {
  int lane = threadIdx.x & 31;
  int wid  = (blockIdx.x * blockDim.x + threadIdx.x) >> 5;
  int m0 = (wid >> 2) * 16;
  int n0 = (wid & 3) * 32;
  int half = lane >> 4, ln = lane & 15;

  const v8bf* A  = (const v8bf*)(ctx + (size_t)(m0 + ln) * M_);
  const v8bf* B0 = (const v8bf*)(cw1T + (size_t)(n0 + ln) * 64);
  const v8bf* B1 = (const v8bf*)(cw1T + (size_t)(n0 + 16 + ln) * 64);
  v8f acc0 = {}, acc1 = {};
#pragma unroll
  for (int kk = 0; kk < 2; ++kk) {
    v16bf a  = SHUF16(A[kk * 4 + half], A[kk * 4 + half + 2]);
    v16bf b0 = SHUF16(B0[kk * 4 + half], B0[kk * 4 + half + 2]);
    v16bf b1 = SHUF16(B1[kk * 4 + half], B1[kk * 4 + half + 2]);
    acc0 = __builtin_amdgcn_wmma_f32_16x16x32_bf16(false, a, false, b0, (short)0, acc0, false, false);
    acc1 = __builtin_amdgcn_wmma_f32_16x16x32_bf16(false, a, false, b1, (short)0, acc1, false, false);
  }
  int c0 = n0 + ln, c1 = n0 + 16 + ln;
  float z0 = cb1[c0], z1 = cb1[c1];
#pragma unroll
  for (int r = 0; r < 8; ++r) {
    int row = m0 + r + 8 * half;
    h2[(size_t)row * 128 + c0] = (__bf16)fmaxf(acc0[r] + z0, 0.f);
    h2[(size_t)row * 128 + c1] = (__bf16)fmaxf(acc1[r] + z1, 0.f);
  }
}

// ================= K5: classifier layer 2 (h2 @ cw2 + cb2) via WMMA =================
__global__ void __launch_bounds__(256)
k5_fc2(const __bf16* __restrict__ h2, const __bf16* __restrict__ cw2T,
       const float* __restrict__ cb2, float* __restrict__ out) {
  int lane = threadIdx.x & 31;
  int wid  = (blockIdx.x * blockDim.x + threadIdx.x) >> 5;
  int m0 = (wid >> 2) * 16;
  int n0 = (wid & 3) * 32;
  int half = lane >> 4, ln = lane & 15;

  const v8bf* A  = (const v8bf*)(h2 + (size_t)(m0 + ln) * 128);
  const v8bf* B0 = (const v8bf*)(cw2T + (size_t)(n0 + ln) * 128);
  const v8bf* B1 = (const v8bf*)(cw2T + (size_t)(n0 + 16 + ln) * 128);
  v8f acc0 = {}, acc1 = {};
#pragma unroll
  for (int kk = 0; kk < 4; ++kk) {
    v16bf a  = SHUF16(A[kk * 4 + half], A[kk * 4 + half + 2]);
    v16bf b0 = SHUF16(B0[kk * 4 + half], B0[kk * 4 + half + 2]);
    v16bf b1 = SHUF16(B1[kk * 4 + half], B1[kk * 4 + half + 2]);
    acc0 = __builtin_amdgcn_wmma_f32_16x16x32_bf16(false, a, false, b0, (short)0, acc0, false, false);
    acc1 = __builtin_amdgcn_wmma_f32_16x16x32_bf16(false, a, false, b1, (short)0, acc1, false, false);
  }
  int c0 = n0 + ln, c1 = n0 + 16 + ln;
#pragma unroll
  for (int r = 0; r < 8; ++r) {
    int row = m0 + r + 8 * half;
    if (c0 < C_) out[(size_t)row * C_ + c0] = acc0[r] + cb2[c0];
    if (c1 < C_) out[(size_t)row * C_ + c1] = acc1[r] + cb2[c1];
  }
}

// ================= host launcher =================
extern "C" void kernel_launch(void* const* d_in, const int* in_sizes, int n_in,
                              void* d_out, int out_size, void* d_ws, size_t ws_size,
                              hipStream_t stream) {
  const float* x     = (const float*)d_in[0];
  const float* ln_g  = (const float*)d_in[1];
  const float* ln_b  = (const float*)d_in[2];
  const float* gleak = (const float*)d_in[3];
  const float* vleak = (const float*)d_in[4];
  const float* cm    = (const float*)d_in[5];
  const float* sigma = (const float*)d_in[6];
  const float* mu    = (const float*)d_in[7];
  const float* w     = (const float*)d_in[8];
  const float* erev  = (const float*)d_in[9];
  const float* s_sg  = (const float*)d_in[10];
  const float* s_mu  = (const float*)d_in[11];
  const float* s_w   = (const float*)d_in[12];
  const float* s_er  = (const float*)d_in[13];
  const float* in_w  = (const float*)d_in[14];
  const float* in_b  = (const float*)d_in[15];
  const float* out_w = (const float*)d_in[16];
  const float* out_b = (const float*)d_in[17];
  const float* aw1   = (const float*)d_in[18];
  const float* ab1   = (const float*)d_in[19];
  const float* aw2   = (const float*)d_in[20];
  const float* cw1   = (const float*)d_in[21];
  const float* cb1   = (const float*)d_in[22];
  const float* cw2   = (const float*)d_in[23];
  const float* cb2   = (const float*)d_in[24];
  const float* mask  = (const float*)d_in[25];
  const float* smask = (const float*)d_in[26];

  char* ws = (char*)d_ws;
  uint2*  syn    = (uint2*)(ws + WS_SYN);
  float4* sens   = (float4*)(ws + WS_SENS);
  float*  rsum   = (float*)(ws + WS_RSUM);
  float*  ssum   = (float*)(ws + WS_SSUM);
  float*  neur   = (float*)(ws + WS_NEUR);
  __bf16* aw1T   = (__bf16*)(ws + WS_AW1T);
  __bf16* cw1T   = (__bf16*)(ws + WS_CW1T);
  __bf16* cw2T   = (__bf16*)(ws + WS_CW2T);
  float*  scores = (float*)(ws + WS_SCORES);
  __bf16* ctx    = (__bf16*)(ws + WS_CTX);
  __bf16* h2     = (__bf16*)(ws + WS_H2);
  __bf16* outs   = (__bf16*)(ws + WS_OUTS);

  // precompute / weight transforms
  k0_syn<<<(U_ * U_ + 255) / 256, 256, 0, stream>>>(sigma, mu, w, erev, mask, syn);
  k0_sens<<<(IN_ * U_ + 255) / 256, 256, 0, stream>>>(s_sg, s_mu, s_w, s_er, smask, sens);
  k0_misc<<<1, 128, 0, stream>>>(w, erev, mask, s_w, s_er, smask, gleak, vleak, cm,
                                 rsum, ssum, neur);
  k0_wt<<<(128 * 128 + 255) / 256, 256, 0, stream>>>(aw1, cw1, cw2, aw1T, cw1T, cw2T);

  // LTC recurrent scan: 128 workgroups x 1024 threads, 148KB LDS each
  k1_ltc<<<B_ / BB, 1024, 0, stream>>>(x, ln_g, ln_b, in_w, in_b, out_w, out_b,
                                       syn, sens, rsum, ssum, neur, outs);

  // attention scores via bf16 WMMA (98304x64 @ 64x32, fused relu + aw2 dot)
  k2_attn<<<(B_ * T_ / 16) / 8, 256, 0, stream>>>(outs, aw1T, ab1, aw2, scores);

  // softmax over time + context pooling
  k3_pool<<<B_, 128, 0, stream>>>(scores, outs, ctx);

  // classifier via bf16 WMMA
  k4_fc1<<<32, 256, 0, stream>>>(ctx, cw1T, cb1, h2);
  k5_fc2<<<32, 256, 0, stream>>>(h2, cw2T, cb2, (float*)d_out);
}
